// DisentangledSelfAttentionWeighted_5334349382088
// MI455X (gfx1250) — matrix operand order repaired and checked by reading, not compiled, not run on hardware
//
#include <hip/hip_runtime.h>
#include <cstdint>
#include <cstddef>

// ---------------------------------------------------------------------------
// CDNA5 (gfx1250) WMMA implementation of DisentangledSelfAttentionWeighted
// B=1024, L=128, E=256, A=256, H=4, D=64
// ---------------------------------------------------------------------------

typedef __attribute__((ext_vector_type(16))) __bf16 v16bf;
typedef __attribute__((ext_vector_type(8)))  float  v8f;
typedef __attribute__((ext_vector_type(4)))  unsigned int v4u;
typedef __attribute__((ext_vector_type(8)))  int    v8i;
typedef __attribute__((ext_vector_type(4)))  int    v4i;

#define DEV __device__ __forceinline__

DEV __bf16 f2bf(float f) { return (__bf16)f; }
DEV float  bf2f(__bf16 x) { return (float)x; }

static constexpr int Bsz  = 1024;
static constexpr int Lql  = 128;
static constexpr int Edim = 256;
static constexpr int Adim = 256;
static constexpr int Hn   = 4;
static constexpr int Dh   = 64;
static constexpr int Mtot = Bsz * Lql; // 131072

// ---------------------------------------------------------------------------
// TDM: load a 128x64 bf16 tile (global row stride 256 elems) into LDS with
// a row pad of 8 bf16 (LDS row stride 72).  D# per CDNA5 ISA ch.8.
// pad: tile row = 64 bf16 = 32 DWORDs -> pad_interval code 4 (32 DW),
//      pad_amount code 3 (4 DWORDs = 16 B = 8 bf16).
// ---------------------------------------------------------------------------
DEV void tdm_load_tile_128x64_bf16(const __bf16* gsrc, __bf16* ldst) {
  unsigned long long ga = (unsigned long long)(uintptr_t)gsrc;
  unsigned lds = (unsigned)(uintptr_t)ldst;   // low 32 bits = LDS byte offset

  v4u g0;
  g0[0] = 1u;                                        // count=1, user descriptor
  g0[1] = lds;                                       // lds_addr
  g0[2] = (unsigned)(ga & 0xffffffffu);              // global_addr[31:0]
  g0[3] = (unsigned)((ga >> 32) & 0x1ffffffu)        // global_addr[56:32]
          | (2u << 30);                              // type = 2 ("image")

  v8i g1;
  g1[0] = (int)((1u << 16)          // data_size = 1 -> 2 bytes
              | (1u << 20)          // pad_enable
              | (4u << 22)          // pad_interval: 32 DWORDs
              | (3u << 25));        // pad_amount: 4 DWORDs
  g1[1] = (int)(256u << 16);        // tensor_dim0[15:0] = 256 (row length)
  g1[2] = (int)(128u << 16);        // tensor_dim0[31:16]=0 | tensor_dim1[15:0]=128
  g1[3] = (int)(64u << 16);         // tensor_dim1[31:16]=0 | tile_dim0 = 64
  g1[4] = (int)(128u);              // tile_dim1 = 128 | tile_dim2 = 0
  g1[5] = (int)(256u);              // tensor_dim0_stride[31:0] = 256 elems
  g1[6] = 0;                        // stride0[47:32]=0 | stride1[15:0]=0
  g1[7] = 0;                        // tensor_dim1_stride[47:16] = 0

  v4i gz4 = {0, 0, 0, 0};           // groups 2/3 unused (2D tensor)
  v8i gz8 = {0, 0, 0, 0, 0, 0, 0, 0};
  __builtin_amdgcn_tensor_load_to_lds(g0, g1, gz4, gz4, gz8, 0);
}

// ---------------------------------------------------------------------------
// Kernel 1: column means over L:  Xbar[b,e] = mean_l X[b,l,e]
// ---------------------------------------------------------------------------
__global__ __launch_bounds__(256) void mean_over_L(const float* __restrict__ X,
                                                   float* __restrict__ Xbar) {
  int idx = blockIdx.x * 256 + threadIdx.x;   // [0, B*E)
  int b = idx >> 8;
  int e = idx & 255;
  const float* p = X + (size_t)b * Lql * Edim + e;
  float s = 0.f;
#pragma unroll 4
  for (int l = 0; l < Lql; ++l) s += p[(size_t)l * Edim];
  Xbar[idx] = s * (1.0f / (float)Lql);
}

// ---------------------------------------------------------------------------
// Kernel 2: tiny GEMM for mean projections: Mu[b,n] = Xbar[b,:] @ W[:,n] + b[n]
// ---------------------------------------------------------------------------
__global__ __launch_bounds__(256) void mu_proj(const float* __restrict__ Xbar,
                                               const float* __restrict__ W,
                                               const float* __restrict__ bias,
                                               float* __restrict__ Mu) {
  int idx = blockIdx.x * 256 + threadIdx.x;   // [0, B*A)
  int b = idx >> 8;
  int n = idx & 255;
  const float* x = Xbar + (size_t)b * Edim;
  float s = bias[n];
#pragma unroll 4
  for (int e = 0; e < Edim; ++e) s += x[e] * W[(size_t)e * Adim + n];
  Mu[idx] = s;
}

// ---------------------------------------------------------------------------
// Kernel 3: projection GEMM  Y[M,256] = X[M,256] @ W[256,256] + bias
// Block tile 64(M) x 256(N = full width) so each activation row is read from
// HBM exactly once per projection.  8 waves as 2(M) x 4(N); each wave owns a
// 32x64 sub-tile = 2x4 WMMA tiles; K-step 32 -> 8 v_wmma per iteration.
// ---------------------------------------------------------------------------
template <typename OUT_T>
__global__ __launch_bounds__(256) void proj_gemm(const float* __restrict__ X,
                                                 const float* __restrict__ W,
                                                 const float* __restrict__ bias,
                                                 OUT_T* __restrict__ Y) {
  constexpr int BK = 32;
  __shared__ __bf16 As[64][BK + 8];     // 64 x 40
  __shared__ __bf16 Bs[BK][Adim + 8];   // 32 x 264

  const int tid  = threadIdx.x;
  const int lane = tid & 31;
  const int half = lane >> 4;
  const int lrow = lane & 15;
  const int wave = tid >> 5;
  const int wm   = wave & 1;     // 2 waves along M (32 rows each)
  const int wn   = wave >> 1;    // 4 waves along N (64 cols each)
  const int m0   = blockIdx.x * 64;

  const v8f zero = {0.f, 0.f, 0.f, 0.f, 0.f, 0.f, 0.f, 0.f};
  v8f acc[2][4];
#pragma unroll
  for (int mt = 0; mt < 2; ++mt)
#pragma unroll
    for (int nt = 0; nt < 4; ++nt) acc[mt][nt] = zero;

  for (int k0 = 0; k0 < Edim; k0 += BK) {
    // stage A tile 64x32 (f32 -> bf16)
#pragma unroll
    for (int i = 0; i < 8; ++i) {
      int e = tid + i * 256;
      int r = e >> 5, c = e & 31;
      As[r][c] = f2bf(X[(size_t)(m0 + r) * Edim + k0 + c]);
    }
    // stage B tile 32x256
#pragma unroll
    for (int i = 0; i < 32; ++i) {
      int e = tid + i * 256;
      int r = e >> 8, c = e & 255;
      Bs[r][c] = f2bf(W[(size_t)(k0 + r) * Adim + c]);
    }
    // prefetch next A K-tile into caches (global_prefetch_b8)
    if (k0 + BK < Edim)
      __builtin_prefetch(X + (size_t)(m0 + (tid & 63)) * Edim + k0 + BK + ((tid >> 6) * 8), 0, 1);
    __syncthreads();

    v16bf a0, a1;
#pragma unroll
    for (int j = 0; j < 16; ++j) {
      int kk = j + half * 16;                         // A 16x32 bf16 layout
      a0[j] = As[wm * 32 + lrow][kk];
      a1[j] = As[wm * 32 + 16 + lrow][kk];
    }
#pragma unroll
    for (int nt = 0; nt < 4; ++nt) {
      v16bf b;
#pragma unroll
      for (int j = 0; j < 16; ++j) {
        int kk = j + half * 16;                       // B 32x16 bf16 layout
        b[j] = Bs[kk][wn * 64 + nt * 16 + lrow];
      }
      acc[0][nt] = __builtin_amdgcn_wmma_f32_16x16x32_bf16(false, a0, false, b,
                                                           (short)0, acc[0][nt], false, false);
      acc[1][nt] = __builtin_amdgcn_wmma_f32_16x16x32_bf16(false, a1, false, b,
                                                           (short)0, acc[1][nt], false, false);
    }
    __syncthreads();
  }

  // epilogue: C/D layout -> global, add bias
#pragma unroll
  for (int mt = 0; mt < 2; ++mt)
#pragma unroll
    for (int nt = 0; nt < 4; ++nt)
#pragma unroll
      for (int r = 0; r < 8; ++r) {
        int row = m0 + wm * 32 + mt * 16 + r + half * 8;
        int col = wn * 64 + nt * 16 + lrow;
        float v = acc[mt][nt][r] + bias[col];
        Y[(size_t)row * Adim + col] = (OUT_T)v;
      }
}

// ---------------------------------------------------------------------------
// Kernel 4: fused attention per (b,h).
//   S = (Q-muQ)(K-muK)^T ; P = softmax_row(S) + softmax(K.muQp) ; O = P@V
//   out[b,l,h*64+d] += O   (residual already resident in out)
// Q/K/V tiles staged by the Tensor Data Mover (tensor_load_to_lds), wave 0,
// completion via s_wait_tensorcnt 0 before the workgroup barrier.
// ---------------------------------------------------------------------------
__global__ __launch_bounds__(256) void attn_kernel(const __bf16* __restrict__ Q,
                                                   const __bf16* __restrict__ K,
                                                   const __bf16* __restrict__ V,
                                                   const float* __restrict__ MuQ,
                                                   const float* __restrict__ MuK,
                                                   const float* __restrict__ MuQp,
                                                   float* __restrict__ Out) {
  extern __shared__ char smem[];
  constexpr int SS = 132;   // f32 stride of S
  constexpr int QS = 72;    // bf16 stride of Q/K/V tiles (TDM pad)
  constexpr int PS = 136;   // bf16 stride of P

  float*  Ss   = (float*)smem;
  float*  un   = Ss + 128 * SS;
  float*  unsm = un + 128;
  float*  muq  = unsm + 128;
  float*  muk  = muq + 64;
  float*  muqp = muk + 64;
  __bf16* Qs   = (__bf16*)(muqp + 64);
  __bf16* Ks   = Qs + 128 * QS;
  __bf16* Vs   = Ks + 128 * QS;
  __bf16* Ps   = Vs + 128 * QS;

  const int tid  = threadIdx.x;
  const int lane = tid & 31;
  const int half = lane >> 4;
  const int lrow = lane & 15;
  const int wave = tid >> 5;
  const int bI   = blockIdx.x >> 2;   // / H
  const int hI   = blockIdx.x & 3;    // % H

  const size_t gbase = ((size_t)bI * Lql) * Adim + (size_t)hI * Dh;

  // TDM staging of Q,K,V tiles (one DMA each, issued by wave 0)
  if (wave == 0) {
    tdm_load_tile_128x64_bf16(Q + gbase, Qs);
    tdm_load_tile_128x64_bf16(K + gbase, Ks);
    tdm_load_tile_128x64_bf16(V + gbase, Vs);
  }

  if (tid < 64)        muq[tid]        = MuQ[(size_t)bI * Adim + hI * Dh + tid];
  else if (tid < 128)  muk[tid - 64]   = MuK[(size_t)bI * Adim + hI * Dh + (tid - 64)];
  else if (tid < 192)  muqp[tid - 128] = MuQp[(size_t)bI * Adim + hI * Dh + (tid - 128)];

  if (wave == 0) __builtin_amdgcn_s_wait_tensorcnt(0);
  __syncthreads();

  // unary raw: un[m] = K[m,:] . muQp   (uncentered K)
  if (tid < 128) {
    float s = 0.f;
#pragma unroll 8
    for (int d = 0; d < 64; ++d) s += bf2f(Ks[tid * QS + d]) * muqp[d];
    un[tid] = s;
  }
  __syncthreads();

  // center Q,K in place; one thread does the tiny unary softmax
#pragma unroll
  for (int i = 0; i < 32; ++i) {
    int e = tid + i * 256;
    int l = e >> 6, d = e & 63;
    Qs[l * QS + d] = f2bf(bf2f(Qs[l * QS + d]) - muq[d]);
    Ks[l * QS + d] = f2bf(bf2f(Ks[l * QS + d]) - muk[d]);
  }
  if (tid == 0) {
    float mx = -3.4e38f;
    for (int n = 0; n < 128; ++n) mx = fmaxf(mx, un[n]);
    float sum = 0.f;
    for (int n = 0; n < 128; ++n) { float e2 = __expf(un[n] - mx); unsm[n] = e2; sum += e2; }
    float inv = 1.f / sum;
    for (int n = 0; n < 128; ++n) unsm[n] *= inv;
  }
  __syncthreads();

  // S = Qc @ Kc^T : wave w owns rows [w*16, w*16+16), all 8 N-tiles
  const int mrow = wave * 16;
  const v8f zero = {0.f, 0.f, 0.f, 0.f, 0.f, 0.f, 0.f, 0.f};
  v8f sacc[8];
#pragma unroll
  for (int i = 0; i < 8; ++i) sacc[i] = zero;

#pragma unroll
  for (int kb = 0; kb < 64; kb += 32) {
    v16bf a;
#pragma unroll
    for (int j = 0; j < 16; ++j) {
      int kk = kb + j + half * 16;
      a[j] = Qs[(mrow + lrow) * QS + kk];
    }
#pragma unroll
    for (int nt = 0; nt < 8; ++nt) {
      v16bf b;
#pragma unroll
      for (int j = 0; j < 16; ++j) {
        int kk = kb + j + half * 16;
        b[j] = Ks[(nt * 16 + lrow) * QS + kk];   // B[k=d][n=key row] = Kc[n][d]
      }
      sacc[nt] = __builtin_amdgcn_wmma_f32_16x16x32_bf16(false, a, false, b,
                                                         (short)0, sacc[nt], false, false);
    }
  }
#pragma unroll
  for (int nt = 0; nt < 8; ++nt)
#pragma unroll
    for (int r = 0; r < 8; ++r)
      Ss[(mrow + r + half * 8) * SS + nt * 16 + lrow] = sacc[nt][r];
  __syncthreads();

  // row softmax + broadcast unary -> Ps (bf16)
  if (tid < 128) {
    const float* srow = Ss + tid * SS;
    float mx = -3.4e38f;
    for (int n = 0; n < 128; ++n) mx = fmaxf(mx, srow[n]);
    float sum = 0.f;
    for (int n = 0; n < 128; ++n) sum += __expf(srow[n] - mx);
    float inv = 1.f / sum;
    __bf16* prow = Ps + tid * PS;
    for (int n = 0; n < 128; ++n)
      prow[n] = f2bf(__expf(srow[n] - mx) * inv + unsm[n]);
  }
  __syncthreads();

  // O = P @ V : wave w rows [w*16, ...), 4 N-tiles over D=64, K over 128 keys
  v8f oacc[4];
#pragma unroll
  for (int i = 0; i < 4; ++i) oacc[i] = zero;

#pragma unroll
  for (int kb = 0; kb < 128; kb += 32) {
    v16bf a;
#pragma unroll
    for (int j = 0; j < 16; ++j) {
      int kk = kb + j + half * 16;
      a[j] = Ps[(mrow + lrow) * PS + kk];
    }
#pragma unroll
    for (int nt = 0; nt < 4; ++nt) {
      v16bf b;
#pragma unroll
      for (int j = 0; j < 16; ++j) {
        int kk = kb + j + half * 16;
        b[j] = Vs[kk * QS + nt * 16 + lrow];     // B[k=key][n=d] = V[key][d]
      }
      oacc[nt] = __builtin_amdgcn_wmma_f32_16x16x32_bf16(false, a, false, b,
                                                         (short)0, oacc[nt], false, false);
    }
  }

  // accumulate onto residual in d_out
#pragma unroll
  for (int nt = 0; nt < 4; ++nt)
#pragma unroll
    for (int r = 0; r < 8; ++r) {
      int l  = mrow + r + half * 8;
      int gn = hI * Dh + nt * 16 + lrow;
      size_t idx = ((size_t)bI * Lql + l) * Adim + gn;
      Out[idx] = Out[idx] + oacc[nt][r];
    }
}

// ---------------------------------------------------------------------------
// Host launcher
// ---------------------------------------------------------------------------
extern "C" void kernel_launch(void* const* d_in, const int* in_sizes, int n_in,
                              void* d_out, int out_size, void* d_ws, size_t ws_size,
                              hipStream_t stream) {
  (void)in_sizes; (void)n_in; (void)out_size; (void)ws_size;

  const float* query = (const float*)d_in[0];
  const float* key   = (const float*)d_in[1];
  const float* value = (const float*)d_in[2];
  const float* Wq = (const float*)d_in[3];
  const float* bq = (const float*)d_in[4];
  const float* Wk = (const float*)d_in[5];
  const float* bk = (const float*)d_in[6];
  const float* Wv = (const float*)d_in[7];
  const float* bv = (const float*)d_in[8];
  const float* Ww = (const float*)d_in[9];
  const float* bw = (const float*)d_in[10];
  const float* Wr = (const float*)d_in[11];
  const float* br = (const float*)d_in[12];
  float* out = (float*)d_out;

  // workspace partition
  char* ws = (char*)d_ws;
  const size_t nQKV = (size_t)Bsz * Lql * Adim;          // 33,554,432 elems
  __bf16* qb = (__bf16*)ws;
  __bf16* kb = qb + nQKV;
  __bf16* vb = kb + nQKV;
  float* fbase = (float*)(ws + 3 * nQKV * sizeof(__bf16));
  float* qbar  = fbase;
  float* kbar  = qbar + (size_t)Bsz * Edim;
  float* mu_q  = kbar + (size_t)Bsz * Edim;
  float* mu_k  = mu_q + (size_t)Bsz * Adim;
  float* mu_qp = mu_k + (size_t)Bsz * Adim;

  dim3 blk(256);

  // 1) L-means of raw inputs
  mean_over_L<<<dim3((Bsz * Edim) / 256), blk, 0, stream>>>(query, qbar);
  mean_over_L<<<dim3((Bsz * Edim) / 256), blk, 0, stream>>>(key,   kbar);

  // 2) mean projections (linearity: mean(x@W+b) = mean(x)@W + b)
  mu_proj<<<dim3((Bsz * Adim) / 256), blk, 0, stream>>>(qbar, Wq, bq, mu_q);
  mu_proj<<<dim3((Bsz * Adim) / 256), blk, 0, stream>>>(kbar, Wk, bk, mu_k);
  mu_proj<<<dim3((Bsz * Adim) / 256), blk, 0, stream>>>(qbar, Ww, bw, mu_qp);

  // 3) WMMA projection GEMMs (q/k/v -> bf16 workspace; residual -> f32 d_out)
  dim3 pg(Mtot / 64);
  proj_gemm<__bf16><<<pg, blk, 0, stream>>>(query, Wq, bq, qb);
  proj_gemm<__bf16><<<pg, blk, 0, stream>>>(key,   Wk, bk, kb);
  proj_gemm<__bf16><<<pg, blk, 0, stream>>>(value, Wv, bv, vb);
  proj_gemm<float ><<<pg, blk, 0, stream>>>(query, Wr, br, out);

  // 4) fused attention, one block per (b,h)
  const size_t smemBytes =
      (size_t)128 * 132 * 4 +            // Ss
      128 * 4 + 128 * 4 + 3 * 64 * 4 +   // un, unsm, mu vectors
      (size_t)3 * 128 * 72 * 2 +         // Qs, Ks, Vs (TDM-padded stride 72)
      (size_t)128 * 136 * 2;             // Ps
  attn_kernel<<<dim3(Bsz * Hn), blk, smemBytes, stream>>>(qb, kb, vb,
                                                          mu_q, mu_k, mu_qp, out);
}
